// Decoder_56530359550871
// MI455X (gfx1250) — compile-verified, matrix operand order checked by reading, not run on hardware
//
#include <hip/hip_runtime.h>
#include <hip/hip_bf16.h>
#include <math.h>

// dims
#define TT 64
#define BB 64
#define SS 256
#define EE 512
#define HH 1024
#define VV 16000
#define K0 (EE + HH + HH)   // 2560: [emb | feed | h_prev0]
#define K1 (HH + HH)        // 2048: [h0n | h_prev1]
#define G4 (4 * HH)         // 4096 gates

typedef __attribute__((ext_vector_type(16))) __bf16 v16bf;
typedef __attribute__((ext_vector_type(8)))  float  v8f;
typedef __attribute__((ext_vector_type(8)))  unsigned short u16x8;
typedef __attribute__((ext_vector_type(4)))  int    v4i;
typedef __attribute__((address_space(1))) v4i gv4i;   // global int4 (b128 payload)
typedef __attribute__((address_space(3))) v4i lv4i;   // LDS int4

#if __has_builtin(__builtin_amdgcn_global_load_async_to_lds_b128) && \
    __has_builtin(__builtin_amdgcn_s_wait_asynccnt)
#define USE_ASYNC 1
#else
#define USE_ASYNC 0
#endif

#if USE_ASYNC
#define ASYNC_WAIT(n) __builtin_amdgcn_s_wait_asynccnt(n)
#else
#define ASYNC_WAIT(n)
#endif

__device__ __forceinline__ unsigned short f2bf(float f) {
  union { float f; unsigned u; } v; v.f = f;
  unsigned r = v.u + 0x7FFFu + ((v.u >> 16) & 1u);   // round-nearest-even
  return (unsigned short)(r >> 16);
}
__device__ __forceinline__ float bf2f(unsigned short h) {
  union { unsigned u; float f; } v; v.u = ((unsigned)h) << 16; return v.f;
}
__device__ __forceinline__ float sigm(float x) { return 1.0f / (1.0f + __expf(-x)); }

// ---------------------------------------------------------------------------
// bf16 WMMA GEMM with LDS-staged, (optionally async) double-buffered tiles.
// C[M x N] = A[M x K](bf16 row-major) @ W[N x K]^T + bias.
// Block: 256 threads = 8 waves; block tile 64x128 = 4 A-tiles x 8 B-tiles.
// Per 32-K step stage 12 tiles (12 KB) into LDS in the exact WMMA per-lane
// layout: tile t, lane l owns 32 contiguous bytes (chunk0 = K half*8..+7,
// chunk1 = K 16+half*8..+7 of row (l&15)). Compute reads are single aligned
// 32B LDS loads; wave w computes m_tile = w&3, n_tiles (w>>2)*4 .. +3.
// grid = (N/128, M/64).
// ---------------------------------------------------------------------------
#define TILE_US 512                       // ushorts per 16x32 bf16 tile
#define NSTAGE_TILES 12                   // 4 A + 8 B
#define STAGE_US (NSTAGE_TILES * TILE_US) // 6144 ushorts = 12 KB

__global__ void gemm_bf16_wmma(const unsigned short* __restrict__ A, int lda,
                               const unsigned short* __restrict__ W, int ldw,
                               const float* __restrict__ bias,
                               float* __restrict__ Cf, int ldcf,
                               unsigned short* __restrict__ Cbf, int ldcb,
                               int K) {
  __shared__ __align__(32) unsigned short sbuf[2][STAGE_US];
  const int tid  = threadIdx.x;
  const int wave = tid >> 5;
  const int lane = tid & 31;
  const int half = lane >> 4;
  const int r    = lane & 15;
  const int mblk = blockIdx.y * 64;
  const int nblk = blockIdx.x * 128;

  // Stage one 32-K step into sbuf[buf]: 768 16B chunks, 3 per thread.
  auto stage = [&](int k, int buf) {
#pragma unroll
    for (int ch = 0; ch < 3; ++ch) {
      int c     = tid + ch * 256;       // chunk id 0..767
      int tile  = c >> 6;               // 0..11
      int q     = c & 63;
      int sl    = q >> 1;               // destination lane 0..31
      int chalf = q & 1;                // which 16B chunk of that lane
      int row   = sl & 15;
      int col   = (sl >> 4) * 8 + chalf * 16;
      const unsigned short* src =
          (tile < 4)
              ? A + (size_t)(mblk + tile * 16 + row) * lda + k + col
              : W + (size_t)(nblk + (tile - 4) * 16 + row) * ldw + k + col;
      unsigned short* dst = &sbuf[buf][tile * TILE_US + sl * 16 + chalf * 8];
#if USE_ASYNC
      __builtin_amdgcn_global_load_async_to_lds_b128(
          (gv4i*)src, (lv4i*)dst, 0, 0);
#else
      u16x8 v = *reinterpret_cast<const u16x8*>(src);
      *reinterpret_cast<u16x8*>(dst) = v;
#endif
    }
  };

  v8f acc[4] = {v8f{}, v8f{}, v8f{}, v8f{}};
  const int nsteps = K >> 5;
  int buf = 0;
  stage(0, 0);
  for (int s = 0; s < nsteps; ++s) {
    if (s + 1 < nsteps) {
      stage((s + 1) << 5, buf ^ 1);     // prefetch next step into other buffer
      ASYNC_WAIT(3);                    // wait for *previous* 3 async issues
    } else {
      ASYNC_WAIT(0);
    }
    __syncthreads();
    const unsigned short* sb = sbuf[buf];
    v16bf a = *reinterpret_cast<const v16bf*>(sb + (wave & 3) * TILE_US + lane * 16);
#pragma unroll
    for (int j = 0; j < 4; ++j) {
      v16bf b = *reinterpret_cast<const v16bf*>(
          sb + (4 + (wave >> 2) * 4 + j) * TILE_US + lane * 16);
      acc[j] = __builtin_amdgcn_wmma_f32_16x16x32_bf16(
          false, a, false, b, (short)0, acc[j], false, false);
    }
    __syncthreads();                    // all reads done before buf is re-staged
    buf ^= 1;
  }

  const int m0 = mblk + (wave & 3) * 16;
  const int n0 = nblk + (wave >> 2) * 64;
#pragma unroll
  for (int j = 0; j < 4; ++j) {
    int n = n0 + j * 16 + r;
    float bv = bias ? bias[n] : 0.0f;
#pragma unroll
    for (int i = 0; i < 8; ++i) {
      int m = m0 + half * 8 + i;        // D layout: VGPR i -> M = half*8+i, N = r
      float v = acc[j][i] + bv;
      if (Cf)  Cf[(size_t)m * ldcf + n] = v;
      if (Cbf) Cbf[(size_t)m * ldcb + n] = f2bf(v);
    }
  }
}

// --------------------------- one-time pack kernels -------------------------
__global__ void pack2_bf16(unsigned short* __restrict__ dst,
                           const float* __restrict__ s1, int k1,
                           const float* __restrict__ s2, int k2, long total) {
  long idx = (long)blockIdx.x * blockDim.x + threadIdx.x;
  if (idx >= total) return;
  int kt = k1 + k2;
  long n = idx / kt; int kk = (int)(idx % kt);
  float v = (kk < k1) ? s1[n * k1 + kk] : s2[n * k2 + (kk - k1)];
  dst[idx] = f2bf(v);
}
__global__ void cvt_bf16(unsigned short* __restrict__ dst,
                         const float* __restrict__ src, long n) {
  long i = (long)blockIdx.x * blockDim.x + threadIdx.x;
  if (i < n) dst[i] = f2bf(src[i]);
}
__global__ void add_bias(float* __restrict__ dst, const float* __restrict__ a,
                         const float* __restrict__ b, int n) {
  int i = blockIdx.x * blockDim.x + threadIdx.x;
  if (i < n) dst[i] = a[i] + b[i];
}
// init recurrent state: c <- c0 ; h_prev regions (bf16) <- h0 ; feed <- 0
__global__ void init_state(const float* __restrict__ h0, const float* __restrict__ c0,
                           float* __restrict__ cbuf,
                           unsigned short* __restrict__ A0,
                           unsigned short* __restrict__ A1) {
  int i = blockIdx.x * blockDim.x + threadIdx.x;       // over 2*B*H
  if (i >= 2 * BB * HH) return;
  cbuf[i] = c0[i];
  int l = i / (BB * HH); int rem = i % (BB * HH);
  int b = rem / HH, j = rem % HH;
  unsigned short hb = f2bf(h0[i]);
  if (l == 0) { A0[(size_t)b * K0 + EE + HH + j] = hb;       // h_prev0
                A0[(size_t)b * K0 + EE + j] = f2bf(0.0f); }  // feed = 0
  else        { A1[(size_t)b * K1 + HH + j] = hb; }          // h_prev1
}

// --------------------------- per-step kernels ------------------------------
__global__ void embed_gather(const int* __restrict__ tok,       // B tokens
                             const float* __restrict__ embed,   // (V,E)
                             unsigned short* __restrict__ A0) {
  int i = blockIdx.x * blockDim.x + threadIdx.x;                // over B*E
  if (i >= BB * EE) return;
  int b = i / EE, e = i % EE;
  A0[(size_t)b * K0 + e] = f2bf(embed[(size_t)tok[b] * EE + e]);
}

// PyTorch gate order i,f,g,o in the 4H axis
__global__ void lstm_pointwise(const float* __restrict__ gates,   // (B,4H)
                               float* __restrict__ c,             // (B,H)
                               unsigned short* __restrict__ hdst0, int ld0,
                               unsigned short* __restrict__ hdst1, int ld1) {
  int i = blockIdx.x * blockDim.x + threadIdx.x;                  // over B*H
  if (i >= BB * HH) return;
  int b = i / HH, j = i % HH;
  const float* g = gates + (size_t)b * G4;
  float ig = sigm(g[j]);
  float fg = sigm(g[HH + j]);
  float gg = tanhf(g[2 * HH + j]);
  float og = sigm(g[3 * HH + j]);
  float cn = fg * c[i] + ig * gg;
  float hn = og * tanhf(cn);
  c[i] = cn;
  unsigned short hb = f2bf(hn);
  hdst0[(size_t)b * ld0 + j] = hb;
  hdst1[(size_t)b * ld1 + j] = hb;
}

// One block per batch b: scores over S, softmax, attention summary.
__global__ void attention(const unsigned short* __restrict__ keys,   // (S,B,H) bf16
                          const unsigned short* __restrict__ attbf,  // (S,B,H) bf16
                          const float* __restrict__ query,           // (B,H) f32
                          const int* __restrict__ att_tokens,        // (S,B)
                          float* __restrict__ distg,                 // (S,B)
                          unsigned short* __restrict__ combA) {      // (B,2H) bf16
  const int b = blockIdx.x, tid = threadIdx.x;
  const int wave = tid >> 5, lane = tid & 31;
  __shared__ float sc[SS];
  __shared__ float red[256];

  for (int s = wave; s < SS; s += 8) {
    const unsigned short* kp = keys + ((size_t)s * BB + b) * HH;
    const float* qp = query + (size_t)b * HH;
    float acc = 0.0f;
    for (int j = lane; j < HH; j += 32) acc += bf2f(kp[j]) * qp[j];
#pragma unroll
    for (int m = 16; m >= 1; m >>= 1) acc += __shfl_xor(acc, m, 32);
    if (lane == 0) {
      bool pad = (att_tokens[s * BB + b] == 0);
      sc[s] = acc + (pad ? -1e9f : 0.0f);
    }
  }
  __syncthreads();

  float v = sc[tid];                       // thread tid owns s = tid (S==256)
  red[tid] = v; __syncthreads();
  for (int st = 128; st >= 1; st >>= 1) {
    if (tid < st) red[tid] = fmaxf(red[tid], red[tid + st]);
    __syncthreads();
  }
  float mx = red[0]; __syncthreads();
  float e = __expf(v - mx);
  red[tid] = e; __syncthreads();
  for (int st = 128; st >= 1; st >>= 1) {
    if (tid < st) red[tid] += red[tid + st];
    __syncthreads();
  }
  float d = e / red[0];
  __syncthreads();
  sc[tid] = d;
  distg[(size_t)tid * BB + b] = d;
  __syncthreads();

  for (int h = tid; h < HH; h += 256) {
    float acc = 0.0f;
    for (int s = 0; s < SS; ++s)
      acc += sc[s] * bf2f(attbf[((size_t)s * BB + b) * HH + h]);
    combA[(size_t)b * K1 + HH + h] = f2bf(acc);   // summary half of [hidden|summary]
  }
}

// One block per row b: softmax over V=16000, write probs+EPS, save probs[COPY]
__global__ void softmax_v(const float* __restrict__ logits,   // (B,V)
                          float* __restrict__ out_t,          // (B,V)
                          float* __restrict__ pcopy) {        // (B)
  const int b = blockIdx.x, tid = threadIdx.x;
  __shared__ float red[256];
  const float* row = logits + (size_t)b * VV;
  float mx = -1e30f;
  for (int v = tid; v < VV; v += 256) mx = fmaxf(mx, row[v]);
  red[tid] = mx; __syncthreads();
  for (int st = 128; st >= 1; st >>= 1) {
    if (tid < st) red[tid] = fmaxf(red[tid], red[tid + st]);
    __syncthreads();
  }
  mx = red[0]; __syncthreads();
  float sum = 0.0f;
  for (int v = tid; v < VV; v += 256) sum += __expf(row[v] - mx);
  red[tid] = sum; __syncthreads();
  for (int st = 128; st >= 1; st >>= 1) {
    if (tid < st) red[tid] += red[tid + st];
    __syncthreads();
  }
  float inv = 1.0f / red[0];
  for (int v = tid; v < VV; v += 256)
    out_t[(size_t)b * VV + v] = __expf(row[v] - mx) * inv + 1e-7f;
  if (tid == 0) pcopy[b] = __expf(row[3] - mx) * inv;   // COPY=3, no EPS
}

__global__ void scatter_copy(const int* __restrict__ att_tokens,   // (S,B)
                             const float* __restrict__ distg,      // (S,B)
                             const float* __restrict__ pcopy,      // (B)
                             float* __restrict__ out_t) {          // (B,V)
  int i = blockIdx.x * blockDim.x + threadIdx.x;                   // over S*B
  if (i >= SS * BB) return;
  int b = i % BB;
  int tok = att_tokens[i];
  atomicAdd(&out_t[(size_t)b * VV + tok], distg[i] * pcopy[b]);
}

__global__ void finalize_log(float* __restrict__ out_t) {          // (B,V)
  long i = (long)blockIdx.x * blockDim.x + threadIdx.x;
  if (i >= (long)BB * VV) return;
  int v = (int)(i % VV);
  float x = (v == 3) ? 0.0f : out_t[i];    // zero COPY column, then log
  out_t[i] = __logf(x);                     // log(0) = -inf matches reference
}

// ---------------------------------------------------------------------------
extern "C" void kernel_launch(void* const* d_in, const int* in_sizes, int n_in,
                              void* d_out, int out_size, void* d_ws, size_t ws_size,
                              hipStream_t stream) {
  const int*   ref_tokens = (const int*)  d_in[0];
  const int*   att_tokens = (const int*)  d_in[1];
  const float* att_feat   = (const float*)d_in[2];
  const float* h0in       = (const float*)d_in[3];
  const float* c0in       = (const float*)d_in[4];
  const float* embed_w    = (const float*)d_in[5];
  const float* W_ih0 = (const float*)d_in[6];
  const float* W_hh0 = (const float*)d_in[7];
  const float* b_ih0 = (const float*)d_in[8];
  const float* b_hh0 = (const float*)d_in[9];
  const float* W_ih1 = (const float*)d_in[10];
  const float* W_hh1 = (const float*)d_in[11];
  const float* b_ih1 = (const float*)d_in[12];
  const float* b_hh1 = (const float*)d_in[13];
  const float* Wk = (const float*)d_in[14];
  const float* bk = (const float*)d_in[15];
  const float* Wq = (const float*)d_in[16];
  const float* bq = (const float*)d_in[17];
  const float* Wc = (const float*)d_in[18];
  const float* bc = (const float*)d_in[19];
  const float* Wp = (const float*)d_in[20];
  const float* bp = (const float*)d_in[21];
  // Wsp/bsp (d_in[22..23]) unused: softmax over a single column == 1.0
  float* out = (float*)d_out;
  char*  ws  = (char*)d_ws;

  size_t off = 0;
  auto carve = [&](size_t bytes) { size_t o = off; off = (off + bytes + 255) & ~(size_t)255; return ws + o; };
  unsigned short* wp0   = (unsigned short*)carve((size_t)G4 * K0 * 2);   // [W_ih0|W_hh0] bf16
  unsigned short* wp1   = (unsigned short*)carve((size_t)G4 * K1 * 2);   // [W_ih1|W_hh1] bf16
  unsigned short* wq_b  = (unsigned short*)carve((size_t)HH * HH * 2);
  unsigned short* wc_b  = (unsigned short*)carve((size_t)HH * 2 * HH * 2);
  unsigned short* wpj_b = (unsigned short*)carve((size_t)VV * HH * 2);   // Wp
  unsigned short* wk_b  = (unsigned short*)carve((size_t)HH * HH * 2);
  float* b0s            = (float*)carve((size_t)G4 * 4);
  float* b1s            = (float*)carve((size_t)G4 * 4);
  unsigned short* attbf = (unsigned short*)carve((size_t)SS * BB * HH * 2);
  unsigned short* keysb = (unsigned short*)carve((size_t)SS * BB * HH * 2);
  unsigned short* A0    = (unsigned short*)carve((size_t)BB * K0 * 2);   // [emb|feed|h0]
  unsigned short* A1    = (unsigned short*)carve((size_t)BB * K1 * 2);   // [h0n|h1]
  unsigned short* combA = (unsigned short*)carve((size_t)BB * K1 * 2);   // [hidden|summary]
  float* gates          = (float*)carve((size_t)BB * G4 * 4);
  float* query          = (float*)carve((size_t)BB * HH * 4);
  float* cbuf           = (float*)carve((size_t)2 * BB * HH * 4);
  float* distg          = (float*)carve((size_t)SS * BB * 4);
  float* pcopy          = (float*)carve((size_t)BB * 4);
  float* logits         = (float*)carve((size_t)BB * VV * 4);
  (void)in_sizes; (void)n_in; (void)out_size; (void)ws_size;

  const int TPB = 256;
  auto blk1 = [](long n) { return dim3((unsigned)((n + 255) / 256)); };

  // ---- one-time setup ----
  pack2_bf16<<<blk1((long)G4 * K0), TPB, 0, stream>>>(wp0, W_ih0, EE + HH, W_hh0, HH, (long)G4 * K0);
  pack2_bf16<<<blk1((long)G4 * K1), TPB, 0, stream>>>(wp1, W_ih1, HH, W_hh1, HH, (long)G4 * K1);
  cvt_bf16<<<blk1((long)HH * HH), TPB, 0, stream>>>(wq_b, Wq, (long)HH * HH);
  cvt_bf16<<<blk1((long)HH * 2 * HH), TPB, 0, stream>>>(wc_b, Wc, (long)HH * 2 * HH);
  cvt_bf16<<<blk1((long)VV * HH), TPB, 0, stream>>>(wpj_b, Wp, (long)VV * HH);
  cvt_bf16<<<blk1((long)HH * HH), TPB, 0, stream>>>(wk_b, Wk, (long)HH * HH);
  cvt_bf16<<<blk1((long)SS * BB * HH), TPB, 0, stream>>>(attbf, att_feat, (long)SS * BB * HH);
  add_bias<<<blk1(G4), TPB, 0, stream>>>(b0s, b_ih0, b_hh0, G4);
  add_bias<<<blk1(G4), TPB, 0, stream>>>(b1s, b_ih1, b_hh1, G4);
  init_state<<<blk1(2 * BB * HH), TPB, 0, stream>>>(h0in, c0in, cbuf, A0, A1);
  // keys = att_features @ Wk^T + bk   (M = S*B = 16384)
  gemm_bf16_wmma<<<dim3(HH / 128, (SS * BB) / 64), TPB, 0, stream>>>(
      attbf, HH, wk_b, HH, bk, nullptr, 0, keysb, HH, HH);

  // ---- decode loop (stream-ordered; single-buffered state) ----
  for (int t = 0; t < TT; ++t) {
    float* out_t = out + (size_t)t * BB * VV;
    embed_gather<<<blk1(BB * EE), TPB, 0, stream>>>(ref_tokens + t * BB, embed_w, A0);
    // layer 0 gates: (B,2560)@(2560,4096)
    gemm_bf16_wmma<<<dim3(G4 / 128, 1), TPB, 0, stream>>>(
        A0, K0, wp0, K0, b0s, gates, G4, nullptr, 0, K0);
    lstm_pointwise<<<blk1(BB * HH), TPB, 0, stream>>>(
        gates, cbuf, A1, K1, A0 + EE + HH, K0);           // h0n -> A1[:, :H], A0 h_prev0
    // layer 1 gates: (B,2048)@(2048,4096)
    gemm_bf16_wmma<<<dim3(G4 / 128, 1), TPB, 0, stream>>>(
        A1, K1, wp1, K1, b1s, gates, G4, nullptr, 0, K1);
    lstm_pointwise<<<blk1(BB * HH), TPB, 0, stream>>>(
        gates, cbuf + BB * HH, combA, K1, A1 + HH, K1);   // hidden -> combA[:, :H], A1 h_prev1
    // query = hidden @ Wq^T + bq   (A = combA, hidden in first H cols)
    gemm_bf16_wmma<<<dim3(HH / 128, 1), TPB, 0, stream>>>(
        combA, K1, wq_b, HH, bq, query, HH, nullptr, 0, HH);
    attention<<<dim3(BB), TPB, 0, stream>>>(keysb, attbf, query, att_tokens, distg, combA);
    // comb = [hidden|summary] @ Wc^T + bc  -> next feed (bf16 directly into A0)
    gemm_bf16_wmma<<<dim3(HH / 128, 1), TPB, 0, stream>>>(
        combA, K1, wc_b, 2 * HH, bc, nullptr, 0, A0 + EE, K0, 2 * HH);
    // logits = comb @ Wp^T + bp   (A = feed region of A0)
    gemm_bf16_wmma<<<dim3(VV / 128, 1), TPB, 0, stream>>>(
        A0 + EE, K0, wpj_b, HH, bp, logits, VV, nullptr, 0, HH);
    softmax_v<<<dim3(BB), TPB, 0, stream>>>(logits, out_t, pcopy);
    scatter_copy<<<blk1(SS * BB), TPB, 0, stream>>>(att_tokens, distg, pcopy, out_t);
    finalize_log<<<blk1((long)BB * VV), TPB, 0, stream>>>(out_t);
  }
}
